// Encoder_58755152609886
// MI455X (gfx1250) — matrix operand order repaired and checked by reading, not compiled, not run on hardware
//
#include <hip/hip_runtime.h>
#include <hip/hip_bf16.h>

typedef __attribute__((ext_vector_type(16))) _Float16 v16h;
typedef __attribute__((ext_vector_type(8)))  _Float16 v8h;
typedef __attribute__((ext_vector_type(8)))  float    v8f;
typedef __attribute__((ext_vector_type(4)))  unsigned int v4u;
typedef __attribute__((ext_vector_type(8)))  int      v8i;
typedef __attribute__((ext_vector_type(4)))  int      v4i;

#define B_   64
#define T_   512
#define E_   256
#define U_   256
#define G_   1024          // 4*U
#define ROWS (B_ * T_)     // 32768

// LDS layout for lstm_scan (dynamic shared, offsets from base 0):
//   [0, 64K)        Zr  f32 [16][1024]
//   [64K, 72K)      h16 f16 [16][256]
//   [72K, 136K)     Zx double buffer, 2 x f16 [16][1024] (32KB each), TDM dest
#define LDS_ZR_OFF   0
#define LDS_H16_OFF  (16 * G_ * 4)
#define LDS_ZX_OFF   (LDS_H16_OFF + 16 * U_ * 2)
#define LDS_ZX_BUF   (16 * G_ * 2)
#define LDS_TOTAL    (LDS_ZX_OFF + 2 * LDS_ZX_BUF)   // 139264 B

// ---------------------------------------------------------------------------
// Kernel 1: convert x -> f16 row-major [ROWS][E], compute mask[row] = any(x!=0)
// ---------------------------------------------------------------------------
__global__ void prep_x_mask(const float* __restrict__ x,
                            _Float16* __restrict__ xh,
                            int* __restrict__ mask) {
  __shared__ int flag;
  const int row = blockIdx.x;
  const int tid = threadIdx.x;        // 256 threads = E
  if (tid == 0) flag = 0;
  __syncthreads();
  const float v = x[(size_t)row * E_ + tid];
  xh[(size_t)row * E_ + tid] = (_Float16)v;
  if (v != 0.0f) flag = 1;            // benign race: all writers store 1
  __syncthreads();
  if (tid == 0) mask[row] = flag;
}

// ---------------------------------------------------------------------------
// Kernel 2: pack weights [K x 1024] f32 row-major into WMMA B-fragment layout.
// B f16 32x16 fragment: n=lane&15; e<8 -> k=(lane>>4)*8+e ; e>=8 -> k=16+(lane>>4)*8+(e-8)
// ---------------------------------------------------------------------------
__global__ void pack_weights(const float* __restrict__ Wk_f,
                             const float* __restrict__ Wk_b,
                             const float* __restrict__ Wr_f,
                             const float* __restrict__ Wr_b,
                             _Float16* __restrict__ wk_p,
                             _Float16* __restrict__ wr_p) {
  const int gid   = blockIdx.x * blockDim.x + threadIdx.x;   // 65536 total
  const int mat   = gid >> 14;           // 16384 entries per matrix
  const int r     = gid & 16383;
  const int ntile = r >> 8;              // 64 n-tiles
  const int kt    = (r >> 5) & 7;        // 8 k-tiles
  const int lane  = r & 31;
  const float* W = (mat == 0) ? Wk_f : (mat == 1) ? Wk_b
                 : (mat == 2) ? Wr_f : Wr_b;
  _Float16* dst = (mat < 2) ? (wk_p + (size_t)mat * (E_ * G_))
                            : (wr_p + (size_t)(mat - 2) * (U_ * G_));
  const int n     = ntile * 16 + (lane & 15);
  const int kbase = kt * 32 + (lane >> 4) * 8;
  const size_t o  = ((size_t)(ntile * 8 + kt) * 32 + lane) * 16;
#pragma unroll
  for (int e = 0; e < 16; ++e)
    dst[o + e] = (_Float16)W[(size_t)(kbase + e + (e >= 8 ? 8 : 0)) * G_ + n];
}

// ---------------------------------------------------------------------------
// A-fragment loader (16-bit A 16x32): lane m = lane&15,
//   elems 0..7  -> K = kt*32 + (lane>>4)*8 + e
//   elems 8..15 -> K = kt*32 + 16 + (lane>>4)*8 + (e-8)
// ---------------------------------------------------------------------------
__device__ __forceinline__ v16h load_a_frag(const _Float16* rowptr, int kt, int hi) {
  const _Float16* p = rowptr + kt * 32 + hi * 8;
  v8h lo = *(const v8h*)(p);
  v8h hh = *(const v8h*)(p + 16);
  return __builtin_shufflevector(lo, hh, 0, 1, 2, 3, 4, 5, 6, 7,
                                         8, 9, 10, 11, 12, 13, 14, 15);
}

// ---------------------------------------------------------------------------
// TDM: async-load one Zx tile (16 rows x 1024 f16, row stride T_*G_ elems)
// from global into LDS at byte offset lds_off. D# per ISA 08_async_tensor §8.
// This toolchain exposes the 6-arg builtin:
//   (uint32x4 g0, int32x8 g1, int32x4 g2, int32x4 g3, int32x8 g4, i32 cpol)
// ---------------------------------------------------------------------------
__device__ __forceinline__ void tdm_load_zx(const _Float16* gptr, unsigned lds_off) {
  const unsigned long long ga = (unsigned long long)(uintptr_t)gptr;
  const unsigned td0 = (unsigned)(T_ * G_);       // tensor_dim0 (elements)
  const unsigned td1 = 16u;                       // tensor_dim1 (rows)
  v4u g0;
  g0[0] = 1u;                                     // count=1, user descriptor
  g0[1] = lds_off;                                // lds_addr (bytes)
  g0[2] = (unsigned)(ga & 0xFFFFFFFFu);           // global_addr[31:0]
  g0[3] = (unsigned)((ga >> 32) & 0x01FFFFFFu)    // global_addr[56:32]
        | (2u << 30);                             // type=2 ("image")
  v8i g1;
  g1[0] = 0x00010000;                             // data_size=1 (2B), wg_mask=0
  g1[1] = (int)((td0 & 0xFFFFu) << 16);           // tensor_dim0[15:0]
  g1[2] = (int)((td0 >> 16) | ((td1 & 0xFFFFu) << 16)); // dim0 hi | dim1 lo
  g1[3] = (int)((td1 >> 16) | (1024u << 16));     // tile_dim0 = 1024
  g1[4] = 16;                                     // tile_dim1=16, tile_dim2=0
  g1[5] = (int)td0;                               // tensor_dim0_stride lo32
  g1[6] = 0;                                      // stride hi | dim1_stride lo
  g1[7] = 0;
  v4i z4 = {0, 0, 0, 0};
  v8i z8 = {0, 0, 0, 0, 0, 0, 0, 0};
  __builtin_amdgcn_tensor_load_to_lds(g0, g1, z4, z4, z8, 0);
}

// ---------------------------------------------------------------------------
// Kernel 3: Zx[dir][row][1024] (f16) = xh[row,:] @ Wk_dir + b_dir
// grid: 2 dirs * 2048 m-blocks; 512 threads = 16 waves, wave handles 4 n-tiles.
// ---------------------------------------------------------------------------
__global__ __launch_bounds__(512)
void xgemm(const _Float16* __restrict__ xh,
           const _Float16* __restrict__ wk_p,
           const float* __restrict__ b_f,
           const float* __restrict__ b_b,
           _Float16* __restrict__ Zx) {
  const int dir    = blockIdx.x >> 11;
  const int mb     = blockIdx.x & 2047;
  const int m_base = mb * 16;
  const int wave   = threadIdx.x >> 5;
  const int lane   = threadIdx.x & 31;
  const int hi     = lane >> 4;
  const _Float16* wp   = wk_p + (size_t)dir * (E_ * G_);
  const float*    bias = dir ? b_b : b_f;
  _Float16*       Z    = Zx + (size_t)dir * ROWS * G_;
  const _Float16* arow = xh + (size_t)(m_base + (lane & 15)) * E_;

  v8f acc[4] = {};
  for (int kt = 0; kt < 8; ++kt) {
    v16h a = load_a_frag(arow, kt, hi);
#pragma unroll
    for (int j = 0; j < 4; ++j) {
      const int ntile = wave * 4 + j;
      v16h b = *(const v16h*)(wp + ((size_t)(ntile * 8 + kt) * 32 + lane) * 16);
      acc[j] = __builtin_amdgcn_wmma_f32_16x16x32_f16(
          false, a, false, b, (short)0, acc[j], false, false);
    }
  }
  // D frag: reg r -> lanes 0-15: (M=r, N=lane); lanes 16-31: (M=8+r, N=lane-16)
#pragma unroll
  for (int j = 0; j < 4; ++j) {
    const int col  = (wave * 4 + j) * 16 + (lane & 15);
    const float bv = bias[col];
    const int rowb = m_base + hi * 8;
#pragma unroll
    for (int r = 0; r < 8; ++r)
      Z[(size_t)(rowb + r) * G_ + col] = (_Float16)(acc[j][r] + bv);
  }
}

// ---------------------------------------------------------------------------
// Kernel 4: persistent bidirectional LSTM scan.
// grid: 8 blocks = 2 dirs * 4 batch-tiles of 16 rows; 512 threads = 16 waves.
// Per step: Zr[16,1024] = h16[16,256] @ Wr (512 WMMAs / WG), then pointwise.
// Zx tile for step s+1 is prefetched into LDS via TDM during step s's WMMAs.
// ---------------------------------------------------------------------------
__global__ __launch_bounds__(512)
void lstm_scan(const _Float16* __restrict__ wr_p,
               const _Float16* __restrict__ Zx,
               const int* __restrict__ mask,
               float* __restrict__ out) {
  extern __shared__ char smem[];
  float*    Zr  = (float*)(smem + LDS_ZR_OFF);     // [16][1024]
  _Float16* h16 = (_Float16*)(smem + LDS_H16_OFF); // [16][256]

  const int dir    = blockIdx.x >> 2;
  const int b_base = (blockIdx.x & 3) * 16;
  const int wave   = threadIdx.x >> 5;
  const int lane   = threadIdx.x & 31;
  const int hi     = lane >> 4;
  const int pm     = threadIdx.x >> 5;          // owned batch row 0..15
  const int pu     = (threadIdx.x & 31) * 8;    // owned unit base

  const _Float16* wp = wr_p + (size_t)dir * (U_ * G_);
  const _Float16* Zd = Zx + (size_t)dir * ROWS * G_;
  const int grow = b_base + pm;

  float c_reg[8];
  float h_reg[8];
#pragma unroll
  for (int j = 0; j < 8; ++j) { c_reg[j] = 0.0f; h_reg[j] = 0.0f; }
  for (int i = threadIdx.x; i < 16 * U_; i += 512) h16[i] = (_Float16)0.0f;

  // TDM prefetch of the first Zx tile (16 rows at t0, row stride T_*G_)
  if (wave == 0) {
    const int t0 = dir ? (T_ - 1) : 0;
    tdm_load_zx(Zd + ((size_t)b_base * T_ + t0) * G_, LDS_ZX_OFF);
  }
  __syncthreads();

  for (int s = 0; s < T_; ++s) {
    const int t = dir ? (T_ - 1 - s) : s;

    // Issue TDM for next step's Zx tile into the other buffer.
    const bool have_next = (s + 1 < T_);
    if (wave == 0 && have_next) {
      const int tn = dir ? (T_ - 2 - s) : (s + 1);
      tdm_load_zx(Zd + ((size_t)b_base * T_ + tn) * G_,
                  LDS_ZX_OFF + ((s + 1) & 1) * LDS_ZX_BUF);
    }

    // ---- Zr = h @ Wr (WMMA) ----
    const _Float16* arow = h16 + (size_t)(lane & 15) * U_;
    v8f acc[4] = {};
#pragma unroll
    for (int kt = 0; kt < 8; ++kt) {
      v16h a = load_a_frag(arow, kt, hi);
#pragma unroll
      for (int j = 0; j < 4; ++j) {
        const int ntile = wave * 4 + j;
        v16h b = *(const v16h*)(wp + ((size_t)(ntile * 8 + kt) * 32 + lane) * 16);
        acc[j] = __builtin_amdgcn_wmma_f32_16x16x32_f16(
            false, a, false, b, (short)0, acc[j], false, false);
      }
    }
#pragma unroll
    for (int j = 0; j < 4; ++j) {
      const int col = (wave * 4 + j) * 16 + (lane & 15);
#pragma unroll
      for (int r = 0; r < 8; ++r)
        Zr[(hi * 8 + r) * G_ + col] = acc[j][r];
    }
    // Wait for the *older* TDM transfer (this step's tile); the next-step
    // transfer may remain in flight (TENSORcnt <= 1).
    if (wave == 0) {
      if (have_next) __builtin_amdgcn_s_wait_tensorcnt(1);
      else           __builtin_amdgcn_s_wait_tensorcnt(0);
    }
    __syncthreads();   // Zr + TDM'd Zx tile visible to all waves

    // ---- pointwise gates (all-tanh LSTM cell) ----
    const _Float16* zx =
        (const _Float16*)(smem + LDS_ZX_OFF + (s & 1) * LDS_ZX_BUF) + pm * G_;
    const int mk = mask[grow * T_ + t];
    v8h zxi = *(const v8h*)(zx + pu);
    v8h zxf = *(const v8h*)(zx + U_ + pu);
    v8h zxg = *(const v8h*)(zx + 2 * U_ + pu);
    v8h zxo = *(const v8h*)(zx + 3 * U_ + pu);
    v8f zri = *(const v8f*)(&Zr[pm * G_ + pu]);
    v8f zrf = *(const v8f*)(&Zr[pm * G_ + U_ + pu]);
    v8f zrg = *(const v8f*)(&Zr[pm * G_ + 2 * U_ + pu]);
    v8f zro = *(const v8f*)(&Zr[pm * G_ + 3 * U_ + pu]);
    float* orow = out + ((size_t)grow * T_ + t) * (2 * U_) + dir * U_ + pu;
    v8h hnew16;
#pragma unroll
    for (int jj = 0; jj < 8; ++jj) {
      const float i_ = tanhf(zri[jj] + (float)zxi[jj]);
      const float f_ = tanhf(zrf[jj] + (float)zxf[jj]);
      const float g_ = tanhf(zrg[jj] + (float)zxg[jj]);
      const float o_ = tanhf(zro[jj] + (float)zxo[jj]);
      const float cn = f_ * c_reg[jj] + i_ * g_;
      const float hn = o_ * tanhf(cn);
      if (mk) { c_reg[jj] = cn; h_reg[jj] = hn; }
      __builtin_nontemporal_store(h_reg[jj], orow + jj);  // streaming output
      hnew16[jj] = (_Float16)h_reg[jj];
    }
    *(v8h*)(&h16[pm * U_ + pu]) = hnew16;   // publish h for next step
    __syncthreads();
  }

  // ---- state_h ----
  const size_t stateOff = (size_t)B_ * T_ * (2 * U_);
#pragma unroll
  for (int jj = 0; jj < 8; ++jj)
    __builtin_nontemporal_store(
        h_reg[jj], out + stateOff + (size_t)grow * (2 * U_) + dir * U_ + pu + jj);
}

// ---------------------------------------------------------------------------
extern "C" void kernel_launch(void* const* d_in, const int* in_sizes, int n_in,
                              void* d_out, int out_size, void* d_ws, size_t ws_size,
                              hipStream_t stream) {
  (void)in_sizes; (void)n_in; (void)out_size; (void)ws_size;
  const float* x    = (const float*)d_in[0];
  const float* Wk_f = (const float*)d_in[1];
  const float* Wr_f = (const float*)d_in[2];
  const float* b_f  = (const float*)d_in[3];
  const float* Wk_b = (const float*)d_in[4];
  const float* Wr_b = (const float*)d_in[5];
  const float* b_b  = (const float*)d_in[6];
  float* out = (float*)d_out;

  char* ws = (char*)d_ws;
  _Float16* xh   = (_Float16*)ws;  ws += (size_t)ROWS * E_ * 2;        // 16 MB
  _Float16* wk_p = (_Float16*)ws;  ws += (size_t)2 * E_ * G_ * 2;      // 1 MB
  _Float16* wr_p = (_Float16*)ws;  ws += (size_t)2 * U_ * G_ * 2;      // 1 MB
  int*      mask = (int*)ws;       ws += (size_t)ROWS * 4;             // 128 KB
  _Float16* Zx   = (_Float16*)ws;  ws += (size_t)2 * ROWS * G_ * 2;    // 128 MB

  prep_x_mask<<<ROWS, 256, 0, stream>>>(x, xh, mask);
  pack_weights<<<256, 256, 0, stream>>>(Wk_f, Wk_b, Wr_f, Wr_b, wk_p, wr_p);
  xgemm<<<4096, 512, 0, stream>>>(xh, wk_p, b_f, b_b, Zx);
  lstm_scan<<<8, 512, LDS_TOTAL, stream>>>(wr_p, Zx, mask, out);
}
